// ResidualAttentionBlock_coarse_68650757259767
// MI455X (gfx1250) — compile-verified, matrix operand order checked by reading, not run on hardware
//
#include <hip/hip_runtime.h>
#include <math.h>

// ---------------------------------------------------------------------------
// Types / helpers
// ---------------------------------------------------------------------------
typedef __attribute__((ext_vector_type(16))) __bf16 v16bf;
typedef __attribute__((ext_vector_type(8)))  float  v8f;
typedef unsigned short u16;   // raw bf16 bits in memory

union FragBF { v16bf v; unsigned u[8]; };

#if defined(__HIP_DEVICE_COMPILE__) && \
    __has_builtin(__builtin_amdgcn_global_load_async_to_lds_b64) && \
    __has_builtin(__builtin_amdgcn_s_wait_asynccnt)
#define HAVE_ASYNC_LDS 1
typedef int v2i_t __attribute__((vector_size(2 * sizeof(int))));
typedef __attribute__((address_space(1))) v2i_t as1_v2i;
typedef __attribute__((address_space(3))) v2i_t as3_v2i;
#else
#define HAVE_ASYNC_LDS 0
#endif

__device__ inline u16 f2bf(float f) {
    union { float f; unsigned u; } c; c.f = f;
    unsigned r = (c.u + 0x7FFFu + ((c.u >> 16) & 1u)) >> 16;
    return (u16)r;
}

__device__ inline v8f wmma_bf16(const FragBF& a, const FragBF& b, v8f c) {
    return __builtin_amdgcn_wmma_f32_16x16x32_bf16(false, a.v, false, b.v,
                                                   (short)0, c, false, false);
}

// A-fragment (16x32 bf16) from row-major src; also works as B-fragment for the
// X @ W^T pattern (W stored (N,K) row-major): lane = M-or-N index, K pairs per
// VGPR: vgpr v holds K = (v>=4?16:0) + 8*half + 2*(v&3) + {0,1}.
__device__ inline void load_frag_nk(FragBF& f, const u16* base, int stride, int lane) {
    const int half = lane >> 4, r = lane & 15;
    const u16* p = base + (size_t)r * stride + half * 8;
#pragma unroll
    for (int v = 0; v < 8; ++v) {
        const int k = ((v >> 2) << 4) + ((v & 3) << 1);
        f.u[v] = *(const unsigned*)(p + k);
    }
}

__device__ inline float blockReduce256(float v, float* red) {
    const int tid = threadIdx.x;
    red[tid] = v; __syncthreads();
#pragma unroll
    for (int s = 128; s > 0; s >>= 1) {
        if (tid < s) red[tid] += red[tid + s];
        __syncthreads();
    }
    float r = red[0]; __syncthreads();
    return r;
}

// ---------------------------------------------------------------------------
// Constants for this problem instance
// ---------------------------------------------------------------------------
#define PP    50
#define BT    384
#define DD    768
#define TOK   (PP * BT)          // 19200 tokens
#define NGRP  32                 // Bt / 12
#define NN    588                // (P-1)*12 tokens per group
#define NPAD  592                // 37 * 16
#define NTIL  37                 // column tiles of 16 (16-bit mask words)
#define LDSS  68                 // padded LDS tile stride (bank-conflict-free)

// ---------------------------------------------------------------------------
// 1) fp32 -> bf16 bulk convert (weights)
// ---------------------------------------------------------------------------
__global__ void k_f2bf(const float* __restrict__ in, u16* __restrict__ out, int n) {
    int i = blockIdx.x * 256 + threadIdx.x;
    if (i < n) out[i] = f2bf(in[i]);
}

// ---------------------------------------------------------------------------
// 2) normalized patches -> bf16, padded to NPAD rows per group
// ---------------------------------------------------------------------------
__global__ void k_prep_pn(const float* __restrict__ x, u16* __restrict__ pn) {
    const int row = blockIdx.x;              // g*NPAD + n
    const int g = row / NPAD, n = row % NPAD;
    const int tid = threadIdx.x;
    u16* dst = pn + (size_t)row * DD;
    __shared__ float red[256];
    if (n >= NN) {
#pragma unroll
        for (int t = 0; t < 3; ++t) dst[tid + t * 256] = 0;
        return;
    }
    const float* s = x + (size_t)(BT + g * NN + n) * DD;   // patches view of x
    float v0[3]; float ss = 0.f;
#pragma unroll
    for (int t = 0; t < 3; ++t) { v0[t] = s[tid + t * 256]; ss += v0[t] * v0[t]; }
    ss = blockReduce256(ss, red);
    const float inv = 1.f / (sqrtf(ss) + 1e-9f);
#pragma unroll
    for (int t = 0; t < 3; ++t) dst[tid + t * 256] = f2bf(v0[t] * inv);
}

// ---------------------------------------------------------------------------
// 3) thresholded similarity bitmask via WMMA: one wave per 16x16 tile
// ---------------------------------------------------------------------------
__global__ void k_sim(const u16* __restrict__ pn, unsigned* __restrict__ sim) {
    const int g = blockIdx.y;
    const int tm = blockIdx.x / NTIL, tn = blockIdx.x % NTIL;
    const int lane = threadIdx.x & 31;
    const u16* base = pn + (size_t)g * NPAD * DD;
    v8f acc = {};
    for (int k = 0; k < DD; k += 32) {
        FragBF a, b;
        load_frag_nk(a, base + (size_t)(tm * 16) * DD + k, DD, lane);
        load_frag_nk(b, base + (size_t)(tn * 16) * DD + k, DD, lane);
        acc = wmma_bf16(a, b, acc);
    }
#pragma unroll
    for (int v = 0; v < 8; ++v) {
        unsigned long long m = __ballot((int)(acc[v] > 0.95f));
        if (lane == 0) {
            const int r0 = tm * 16 + v;
            sim[((size_t)g * NPAD + r0) * NTIL + tn]     = (unsigned)(m & 0xFFFFu);
            sim[((size_t)g * NPAD + r0 + 8) * NTIL + tn] = (unsigned)((m >> 16) & 0xFFFFu);
        }
    }
}

// ---------------------------------------------------------------------------
// 4) greedy hub clustering -> per-row assignment (cm is row-one-hot)
// ---------------------------------------------------------------------------
__global__ void k_cluster(const unsigned* __restrict__ sim, int* __restrict__ assignG) {
    const int g = blockIdx.x, tid = threadIdx.x;
    __shared__ int counts[NPAD];
    __shared__ int psOff[NPAD];
    __shared__ int asg[NPAD];
    __shared__ unsigned char zf[NPAD];
    __shared__ int total, rsh;
    __shared__ int bc[256], bi[256];

    int lt = 0;
    for (int n = tid; n < NPAD; n += 256) {
        if (n < NN) {
            const unsigned* row = sim + ((size_t)g * NPAD + n) * NTIL;
            int c = 0;
#pragma unroll
            for (int t = 0; t < NTIL; ++t) c += __popc(row[t]);
            counts[n] = c;
            const int dg = (row[n >> 4] >> (n & 15)) & 1;
            psOff[n] = c - dg; zf[n] = 0; lt += c - dg;
        } else { counts[n] = -1; psOff[n] = 0; zf[n] = 1; }
        asg[n] = n;
    }
    if (tid == 0) total = 0;
    __syncthreads();
    atomicAdd(&total, lt);
    __syncthreads();

    int iter = 0;
    while (true) {
        int bcv = -2, biv = 0;
        for (int n = tid; n < NPAD; n += 256) {
            const int c = counts[n];
            if (c > bcv) { bcv = c; biv = n; }
        }
        bc[tid] = bcv; bi[tid] = biv;
        __syncthreads();
        if (tid == 0) {
            int c = -2, ii = 0;
            for (int t = 0; t < 256; ++t)
                if (bc[t] > c || (bc[t] == c && bi[t] < ii)) { c = bc[t]; ii = bi[t]; }
            rsh = ii;
        }
        __syncthreads();
        const int r = rsh;
        const unsigned* rrow = sim + ((size_t)g * NPAD + r) * NTIL;  // hub never a prior member
        for (int n = tid; n < NPAD; n += 256) {
            if ((rrow[n >> 4] >> (n & 15)) & 1) {
                if (!zf[n]) { atomicSub(&total, psOff[n]); zf[n] = 1; }
                asg[n] = r;
                counts[n] = 1;
            }
        }
        __syncthreads();
        ++iter;
        if (!(total >= 1 && iter < 1000)) break;
    }
    __syncthreads();
    for (int n = tid; n < NN; n += 256) assignG[g * NPAD + n] = asg[n];
}

// ---------------------------------------------------------------------------
// 5) xin = concat(x[:1], gather(patches, assign))   (fp32)
// ---------------------------------------------------------------------------
__global__ void k_gather(const float* __restrict__ x, const int* __restrict__ assignG,
                         float* __restrict__ xin) {
    const int row = blockIdx.x, tid = threadIdx.x;
    size_t src;
    if (row < BT) src = row;
    else {
        const int tt = row - BT;
        const int g = tt / NN, n = tt % NN;
        src = (size_t)BT + (size_t)g * NN + assignG[g * NPAD + n];
    }
    const float* s = x + src * (size_t)DD;
    float* d = xin + (size_t)row * DD;
#pragma unroll
    for (int t = 0; t < 3; ++t) d[tid + t * 256] = s[tid + t * 256];
}

// ---------------------------------------------------------------------------
// 6) LayerNorm (fp32 in -> bf16 out), one block per token
// ---------------------------------------------------------------------------
__global__ void k_layernorm(const float* __restrict__ X, const float* __restrict__ gam,
                            const float* __restrict__ bet, u16* __restrict__ out) {
    const int row = blockIdx.x, tid = threadIdx.x;
    const float* xr = X + (size_t)row * DD;
    __shared__ float red[256];
    float v0[3]; float s = 0.f;
#pragma unroll
    for (int t = 0; t < 3; ++t) { v0[t] = xr[tid + t * 256]; s += v0[t]; }
    s = blockReduce256(s, red);
    const float mean = s * (1.f / DD);
    float s2 = 0.f;
#pragma unroll
    for (int t = 0; t < 3; ++t) { const float d = v0[t] - mean; s2 += d * d; }
    s2 = blockReduce256(s2, red);
    const float inv = rsqrtf(s2 * (1.f / DD) + 1e-5f);
#pragma unroll
    for (int t = 0; t < 3; ++t) {
        const int d = tid + t * 256;
        out[(size_t)row * DD + d] = f2bf((v0[t] - mean) * inv * gam[d] + bet[d]);
    }
}

// ---------------------------------------------------------------------------
// 7) GEMM: out(M,N) = A(M,K)bf16 @ W(N,K)bf16^T + bias, with epilogues.
//    Block = 2 waves; each wave computes 64(M) x 64(N): 16 WMMA accumulators,
//    4 A-frags reused across 4 B-frags -> ~43 FLOP per L2 byte.
//    MODE 0: store bf16          MODE 1: += residual, store fp32
//    MODE 2: QuickGELU, bf16     MODE 3: += residual, store fp32 (final)
// ---------------------------------------------------------------------------
template <int MODE>
__global__ void k_gemm(const u16* __restrict__ A, const u16* __restrict__ W,
                       const float* __restrict__ bias, const float* __restrict__ res,
                       float* __restrict__ outF, u16* __restrict__ outB,
                       int N, int K) {
    const int wave = threadIdx.x >> 5, lane = threadIdx.x & 31;
    const int half = lane >> 4, lm = lane & 15;
    const size_t m0 = (size_t)blockIdx.y * 128 + wave * 64;
    const int n0 = blockIdx.x * 64;
    v8f acc[4][4];
#pragma unroll
    for (int i = 0; i < 4; ++i)
#pragma unroll
        for (int j = 0; j < 4; ++j) acc[i][j] = {};
    const u16* Ar[4];
#pragma unroll
    for (int i = 0; i < 4; ++i) Ar[i] = A + (m0 + i * 16) * K;

    for (int k = 0; k < K; k += 32) {
        FragBF a[4];
#pragma unroll
        for (int i = 0; i < 4; ++i) {
            load_frag_nk(a[i], Ar[i] + k, K, lane);
            __builtin_prefetch((const void*)(Ar[i] + k + 64), 0, 1);  // global_prefetch_b8
        }
#pragma unroll
        for (int j = 0; j < 4; ++j) {
            const u16* wp = W + (size_t)(n0 + j * 16) * K + k;
            FragBF bb;
            load_frag_nk(bb, wp, K, lane);
            __builtin_prefetch((const void*)(wp + 64), 0, 1);
            acc[0][j] = wmma_bf16(a[0], bb, acc[0][j]);
            acc[1][j] = wmma_bf16(a[1], bb, acc[1][j]);
            acc[2][j] = wmma_bf16(a[2], bb, acc[2][j]);
            acc[3][j] = wmma_bf16(a[3], bb, acc[3][j]);
        }
    }
#pragma unroll
    for (int i2 = 0; i2 < 4; ++i2)
#pragma unroll
        for (int j = 0; j < 4; ++j)
#pragma unroll
            for (int v = 0; v < 8; ++v) {
                const size_t m = m0 + i2 * 16 + v + 8 * half;
                const int n = n0 + j * 16 + lm;
                float val = acc[i2][j][v] + bias[n];
                const size_t idx = m * (size_t)N + n;
                if (MODE == 0) outB[idx] = f2bf(val);
                else if (MODE == 1) outF[idx] = val + res[idx];
                else if (MODE == 2) {
                    const float gl = val / (1.f + __expf(-1.702f * val));
                    outB[idx] = f2bf(gl);
                } else outF[idx] = val + res[idx];
            }
}

// ---------------------------------------------------------------------------
// 8) Attention: one workgroup (4 waves) per (b, h). Q/K staged via async
//    LDS copies when available; V stored transposed (conflict-free with
//    stride-68 padding). WMMA scores -> register softmax -> WMMA att @ V.
// ---------------------------------------------------------------------------
__global__ void k_attention(const u16* __restrict__ qkv, u16* __restrict__ outb) {
    const int b = blockIdx.x % BT;
    const int h = blockIdx.x / BT;
    __shared__ __align__(16) u16 Qs[64 * LDSS];
    __shared__ __align__(16) u16 Ks[64 * LDSS];
    __shared__ __align__(16) u16 Vt[64 * LDSS];   // transposed: Vt[e][l]
    __shared__ __align__(16) u16 As[64 * LDSS];
    const int tid = threadIdx.x;

#if HAVE_ASYNC_LDS
    // zero the padding (rows >= 50 of Q/K, cols >= 50 of Vt)
    for (int idx = tid; idx < 14 * 64; idx += 128) {
        const int l = 50 + (idx >> 6), e = idx & 63;
        Qs[l * LDSS + e] = 0; Ks[l * LDSS + e] = 0; Vt[e * LDSS + l] = 0;
    }
    // async copy Q/K rows (b64 chunks: 4 bf16 each, 8B-aligned in LDS)
    for (int idx = tid; idx < 50 * 16; idx += 128) {
        const int l = idx >> 4, c = idx & 15;
        const size_t gb = ((size_t)(l * BT + b)) * (3 * DD) + h * 64 + c * 4;
        const int lo = l * LDSS + c * 4;
        __builtin_amdgcn_global_load_async_to_lds_b64(
            (as1_v2i*)(qkv + gb), (as3_v2i*)&Qs[lo], 0, 0);
        __builtin_amdgcn_global_load_async_to_lds_b64(
            (as1_v2i*)(qkv + gb + DD), (as3_v2i*)&Ks[lo], 0, 0);
    }
    // V: transposed store (cannot async-transpose)
    for (int idx = tid; idx < 50 * 64; idx += 128) {
        const int l = idx >> 6, e = idx & 63;
        Vt[e * LDSS + l] = qkv[((size_t)(l * BT + b)) * (3 * DD) + 2 * DD + h * 64 + e];
    }
    __builtin_amdgcn_s_wait_asynccnt(0);
    __syncthreads();
#else
    for (int idx = tid; idx < 4096; idx += 128) {
        const int l = idx >> 6, e = idx & 63;
        u16 q = 0, kk = 0, vv = 0;
        if (l < PP) {
            const size_t base = ((size_t)(l * BT + b)) * (3 * DD) + h * 64 + e;
            q = qkv[base]; kk = qkv[base + DD]; vv = qkv[base + 2 * DD];
        }
        Qs[l * LDSS + e] = q; Ks[l * LDSS + e] = kk; Vt[e * LDSS + l] = vv;
    }
    __syncthreads();
#endif

    const int w = tid >> 5, lane = tid & 31, half = lane >> 4, lm = lane & 15;
    v8f acc[4];
#pragma unroll
    for (int j = 0; j < 4; ++j) acc[j] = {};
    for (int k0 = 0; k0 < 64; k0 += 32) {
        FragBF a;
        load_frag_nk(a, Qs + (w * 16) * LDSS + k0, LDSS, lane);
#pragma unroll
        for (int j = 0; j < 4; ++j) {
            FragBF bb;
            load_frag_nk(bb, Ks + (j * 16) * LDSS + k0, LDSS, lane);
            acc[j] = wmma_bf16(a, bb, acc[j]);
        }
    }
    // softmax over the 50 valid key columns, rows live in registers
#pragma unroll
    for (int v = 0; v < 8; ++v) {
        float mx = -1e30f;
#pragma unroll
        for (int j = 0; j < 4; ++j) {
            const float val = acc[j][v] * 0.125f;   // 1/sqrt(64)
            acc[j][v] = val;
            const bool valid = (j * 16 + lm) < PP;
            mx = fmaxf(mx, valid ? val : -1e30f);
        }
#pragma unroll
        for (int off = 1; off < 16; off <<= 1) mx = fmaxf(mx, __shfl_xor(mx, off));
        float s = 0.f;
#pragma unroll
        for (int j = 0; j < 4; ++j) {
            const bool valid = (j * 16 + lm) < PP;
            const float e = valid ? __expf(acc[j][v] - mx) : 0.f;
            acc[j][v] = e; s += e;
        }
#pragma unroll
        for (int off = 1; off < 16; off <<= 1) s += __shfl_xor(s, off);
        const float inv = 1.f / s;
        const int m = w * 16 + v + 8 * half;
#pragma unroll
        for (int j = 0; j < 4; ++j) As[m * LDSS + j * 16 + lm] = f2bf(acc[j][v] * inv);
    }
    __syncthreads();

    v8f o[4];
#pragma unroll
    for (int j = 0; j < 4; ++j) o[j] = {};
    for (int k0 = 0; k0 < 64; k0 += 32) {
        FragBF a;
        load_frag_nk(a, As + (w * 16) * LDSS + k0, LDSS, lane);
#pragma unroll
        for (int j = 0; j < 4; ++j) {
            FragBF bb;   // B[k=l][n=e] = Vt[e][l] -> same contiguous-pair loader
            load_frag_nk(bb, Vt + (j * 16) * LDSS + k0, LDSS, lane);
            o[j] = wmma_bf16(a, bb, o[j]);
        }
    }
#pragma unroll
    for (int j = 0; j < 4; ++j)
#pragma unroll
        for (int v = 0; v < 8; ++v) {
            const int m = w * 16 + v + 8 * half;
            if (m < PP)
                outb[((size_t)(m * BT + b)) * DD + h * 64 + j * 16 + lm] = f2bf(o[j][v]);
        }
}

// ---------------------------------------------------------------------------
// Launch
// ---------------------------------------------------------------------------
extern "C" void kernel_launch(void* const* d_in, const int* in_sizes, int n_in,
                              void* d_out, int out_size, void* d_ws, size_t ws_size,
                              hipStream_t stream) {
    (void)in_sizes; (void)n_in; (void)out_size; (void)ws_size;
    const float* x      = (const float*)d_in[0];
    const float* ln1_g  = (const float*)d_in[1];
    const float* ln1_b  = (const float*)d_in[2];
    const float* in_w   = (const float*)d_in[3];
    const float* in_b   = (const float*)d_in[4];
    const float* out_w  = (const float*)d_in[5];
    const float* out_b  = (const float*)d_in[6];
    const float* ln2_g  = (const float*)d_in[7];
    const float* ln2_b  = (const float*)d_in[8];
    const float* fc_w   = (const float*)d_in[9];
    const float* fc_b   = (const float*)d_in[10];
    const float* proj_w = (const float*)d_in[11];
    const float* proj_b = (const float*)d_in[12];
    float* out = (float*)d_out;

    char* ws = (char*)d_ws;
    size_t off = 0;
    auto alloc = [&](size_t bytes) -> void* {
        void* p = ws + off;
        off += (bytes + 255) & ~(size_t)255;
        return p;
    };
    u16*      wq    = (u16*)alloc((size_t)3 * DD * DD * 2);           // in_w bf16  (2304x768)
    u16*      wo    = (u16*)alloc((size_t)DD * DD * 2);               // out_w bf16
    u16*      wf    = (u16*)alloc((size_t)4 * DD * DD * 2);           // fc_w bf16  (3072x768)
    u16*      wp    = (u16*)alloc((size_t)DD * 4 * DD * 2);           // proj_w bf16 (768x3072)
    unsigned* simm  = (unsigned*)alloc((size_t)NGRP * NPAD * NTIL * 4);
    int*      asg   = (int*)alloc((size_t)NGRP * NPAD * 4);
    float*    xin   = (float*)alloc((size_t)TOK * DD * 4);
    u16*      ln1o  = (u16*)alloc((size_t)TOK * DD * 2);
    u16*      big   = (u16*)alloc((size_t)TOK * 4 * DD * 2);          // qkv_bf / hact (aliased)
    u16*      attnb = (u16*)alloc((size_t)TOK * DD * 2);
    float*    x2    = (float*)alloc((size_t)TOK * DD * 4);
    u16*      pnln2 = (u16*)alloc((size_t)TOK * DD * 2);              // pn / ln2_bf (aliased)
    u16* qkvb = big;  u16* hact = big;
    u16* pn   = pnln2; u16* ln2o = pnln2;

    // weights -> bf16
    k_f2bf<<<(3 * DD * DD + 255) / 256, 256, 0, stream>>>(in_w,   wq, 3 * DD * DD);
    k_f2bf<<<(DD * DD + 255) / 256, 256, 0, stream>>>(out_w,  wo, DD * DD);
    k_f2bf<<<(4 * DD * DD + 255) / 256, 256, 0, stream>>>(fc_w,   wf, 4 * DD * DD);
    k_f2bf<<<(4 * DD * DD + 255) / 256, 256, 0, stream>>>(proj_w, wp, 4 * DD * DD);

    // clustering path (video_frame = 12 >= 1 always for this harness)
    k_prep_pn<<<NGRP * NPAD, 256, 0, stream>>>(x, pn);
    k_sim<<<dim3(NTIL * NTIL, NGRP), 32, 0, stream>>>(pn, simm);
    k_cluster<<<NGRP, 256, 0, stream>>>(simm, asg);
    k_gather<<<TOK, 256, 0, stream>>>(x, asg, xin);

    // attention block
    k_layernorm<<<TOK, 256, 0, stream>>>(xin, ln1_g, ln1_b, ln1o);
    k_gemm<0><<<dim3(3 * DD / 64, TOK / 128), 64, 0, stream>>>(ln1o, wq, in_b, nullptr,
                                                               nullptr, qkvb, 3 * DD, DD);
    k_attention<<<BT * 12, 128, 0, stream>>>(qkvb, attnb);
    k_gemm<1><<<dim3(DD / 64, TOK / 128), 64, 0, stream>>>(attnb, wo, out_b, xin,
                                                           x2, nullptr, DD, DD);
    // MLP
    k_layernorm<<<TOK, 256, 0, stream>>>(x2, ln2_g, ln2_b, ln2o);
    k_gemm<2><<<dim3(4 * DD / 64, TOK / 128), 64, 0, stream>>>(ln2o, wf, fc_b, nullptr,
                                                               nullptr, hact, 4 * DD, DD);
    k_gemm<3><<<dim3(DD / 64, TOK / 128), 64, 0, stream>>>(hact, wp, proj_b, x2,
                                                           out, nullptr, DD, 4 * DD);
}
